// HEGNN_27384711479754
// MI455X (gfx1250) — matrix operand order repaired and checked by reading, not compiled, not run on hardware
//
#include <hip/hip_runtime.h>
#include <math.h>

// ---------------- Problem constants ----------------
static constexpr int   kN   = 20000;
static constexpr int   kE   = 640000;
static constexpr float kCUT = 2.0f;
static constexpr float kEPS = 1e-8f;
static constexpr float kPI  = 3.14159265358979323846f;

typedef __attribute__((ext_vector_type(16))) _Float16 v16h;
typedef __attribute__((ext_vector_type(8)))  _Float16 v8h;
typedef __attribute__((ext_vector_type(8)))  float    v8f;

// sigmoid denominator >= 1 -> raw v_rcp_f32 is safe; avoids IEEE div sequence
__device__ __forceinline__ float silu_f(float x) {
  return x * __builtin_amdgcn_rcpf(1.0f + __expf(-x));
}

// ---------------------------------------------------------------------------
// WMMA helpers (CDNA5 / gfx1250, wave32). Layouts (K order contiguous in A and B, not the ISA table order):
//  A (16x32 f16): row M = rowBase + lane%16; half i holds
//                 K = kBase + i + (lane>=16 ? 16 : 0)
//  B (32x16 f16): col N = colBase + lane%16; half i holds
//                 K = kBase + i + (lane>=16 ? 16 : 0)
//  C/D (16x16 f32): vgpr v: M = rowBase + v + (lane>=16 ? 8 : 0), N = colBase+lane%16
//
// B operands come from a *fragment-swizzled* f16 pack: for K-tile kt, column n,
// half-wave kh, the 16 halves a lane needs live contiguously at
//   Wswz[((kt*64 + n)*2 + kh)*16 .. +15]
// so one lane fragment = one aligned 32-byte load (2x global_load_b128).
// ---------------------------------------------------------------------------
template <int KT>
__device__ __forceinline__ v8f wave_gemm_f16_acc(v8f c, const _Float16* __restrict__ X, int ldk,
                                                 int rowBase,
                                                 const _Float16* __restrict__ Wswz, int ktOff,
                                                 int colBase, int lane) {
#pragma unroll
  for (int kt = 0; kt < KT; ++kt) {
    v16h a;
    {
      const int m = rowBase + (lane & 15);
      const _Float16* xr = X + (size_t)m * ldk + kt * 32 + ((lane & 16) ? 16 : 0);
#pragma unroll
      for (int i = 0; i < 16; ++i) { a[i] = xr[i]; }
    }
    const v16h b = *(const v16h*)(Wswz +
                                  ((size_t)((ktOff + kt) * 64 + colBase + (lane & 15)) << 5) +
                                  ((lane & 16) ? 16 : 0));
    c = __builtin_amdgcn_wmma_f32_16x16x32_f16(false, a, false, b, (short)0, c, false, false);
  }
  return c;
}

template <int KT>
__device__ __forceinline__ v8f wave_gemm_f16(const _Float16* __restrict__ X, int ldk, int rowBase,
                                             const _Float16* __restrict__ Wswz, int colBase,
                                             int lane) {
  v8f c = {};
  return wave_gemm_f16_acc<KT>(c, X, ldk, rowBase, Wswz, 0, colBase, lane);
}

__device__ __forceinline__ void store_silu_lds(_Float16* Y, int ldn, int rowBase, int colBase,
                                               int lane, v8f c, const float* __restrict__ bias) {
  const int n  = colBase + (lane & 15);
  const int mo = (lane & 16) ? 8 : 0;
  const float bv = bias[n];
#pragma unroll
  for (int v = 0; v < 8; ++v)
    Y[(size_t)(rowBase + mo + v) * ldn + n] = (_Float16)silu_f(c[v] + bv);
}

// Bessel radial basis with polynomial envelope (PENV=5): sqrt(2/CUT)=1 for CUT=2.
__device__ __forceinline__ void radial_basis(float rr, float* out16) {
  const float d = rr * (1.0f / kCUT);
  float env = 0.0f;
  if (d < 1.0f) {
    const float d5 = d * d * d * d * d;
    env = 1.0f - 21.0f * d5 + 35.0f * d5 * d - 15.0f * d5 * d * d;
  }
  const float s = env / (rr + kEPS);
#pragma unroll
  for (int n = 0; n < 16; ++n) out16[n] = __sinf((float)(n + 1) * kPI * d) * s;
}

__device__ __forceinline__ void sh_basis(float x, float y, float z, float* s9) {
  const float s3 = 1.7320508075688772f, s15 = 3.872983346207417f;
  const float s5h = 1.118033988749895f, s15h = 1.9364916731037085f;
  s9[0] = 1.0f;
  s9[1] = s3 * x; s9[2] = s3 * y; s9[3] = s3 * z;
  s9[4] = s15 * x * y; s9[5] = s15 * y * z;
  s9[6] = s5h * (2.0f * z * z - x * x - y * y);
  s9[7] = s15 * x * z; s9[8] = s15h * (x * x - y * y);
}

// ---------------------------------------------------------------------------
// Utility kernels
// ---------------------------------------------------------------------------
__global__ void zero_kernel(float* p, int n) {
  int i = blockIdx.x * blockDim.x + threadIdx.x;
  if (i < n) p[i] = 0.0f;
}

// fp32 (K x 64) -> fragment-swizzled f16 pack (KT K-tiles of 32, zero padded).
__global__ void convert_w_kernel(const float* __restrict__ src, _Float16* __restrict__ dst,
                                 int K, int KT) {
  int i = blockIdx.x * blockDim.x + threadIdx.x;
  if (i >= KT * 2048) return;
  int kt = i >> 11;
  int n  = (i >> 5) & 63;
  int k  = kt * 32 + (i & 31);
  dst[i] = (k < K) ? (_Float16)src[(size_t)k * 64 + n] : (_Float16)0.0f;
}

// fp32 (64 x ncols), ncols<=16 -> KT=2 pack with cols [0,ncols) used.
__global__ void convert_small_w_kernel(const float* __restrict__ src, _Float16* __restrict__ dst,
                                       int ncols) {
  int i = blockIdx.x * blockDim.x + threadIdx.x;
  if (i >= 2 * 2048) return;
  int kt = i >> 11;
  int n  = (i >> 5) & 63;
  int k  = kt * 32 + (i & 31);
  dst[i] = (n < ncols) ? (_Float16)src[(size_t)k * ncols + n] : (_Float16)0.0f;
}

// Block-diagonal 192x16 gate pack (KT=6): K 0-63 pos W2 -> cols 0-1,
// K 64-127 vel W2 -> cols 2-3, K 128-191 sh W2 -> cols 4-6.
__global__ void convert_gate_w_kernel(const float* __restrict__ posW2,
                                      const float* __restrict__ velW2,
                                      const float* __restrict__ shW2,
                                      _Float16* __restrict__ dst) {
  int i = blockIdx.x * blockDim.x + threadIdx.x;
  if (i >= 6 * 2048) return;
  int kt = i >> 11;
  int n  = (i >> 5) & 63;
  int k  = kt * 32 + (i & 31);
  float v = 0.0f;
  if (k < 64)       { if (n < 2) v = posW2[(size_t)k * 2 + n]; }
  else if (k < 128) { if (n >= 2 && n < 4) v = velW2[(size_t)(k - 64) * 2 + (n - 2)]; }
  else              { if (n >= 4 && n < 7) v = shW2[(size_t)(k - 128) * 3 + (n - 4)]; }
  dst[i] = (_Float16)v;
}

__global__ void embed_kernel(const float* __restrict__ nf, const float* __restrict__ W,
                             const float* __restrict__ b, _Float16* __restrict__ h16) {
  int i = blockIdx.x * blockDim.x + threadIdx.x;
  if (i >= kN * 64) return;
  int n = i >> 6, c = i & 63;
  float acc = b[c];
#pragma unroll
  for (int k = 0; k < 4; ++k) acc += nf[n * 4 + k] * W[k * 64 + c];
  h16[i] = (_Float16)acc;
}

__global__ void edge_geom_kernel(const float* __restrict__ pos, const float* __restrict__ vel,
                                 const int* __restrict__ eidx, float* __restrict__ relv,
                                 float* __restrict__ dvelv, float* __restrict__ deg) {
  int e = blockIdx.x * blockDim.x + threadIdx.x;
  if (e >= kE) return;
  int r = eidx[e], c = eidx[kE + e];
#pragma unroll
  for (int k = 0; k < 3; ++k) {
    relv[e * 3 + k]  = pos[r * 3 + k] - pos[c * 3 + k];
    dvelv[e * 3 + k] = vel[r * 3 + k] - vel[c * 3 + k];
  }
  atomicAdd(&deg[r], 1.0f);
}

__global__ void finalize_sh_kernel(const float* __restrict__ sh_sum,
                                   const float* __restrict__ deg, float* __restrict__ node_sh) {
  int i = blockIdx.x * blockDim.x + threadIdx.x;
  if (i >= kN * 9) return;
  node_sh[i] = sh_sum[i] / fmaxf(deg[i / 9], 1.0f);
}

// ---------------------------------------------------------------------------
// sh_init: g0 = MLP(144->64->3)(h[row],h[col],radial); scatter expand(g0)*sh_e
// ---------------------------------------------------------------------------
struct ShInitArgs {
  const int* eidx; const _Float16* h16; const float* relv;
  const _Float16* W1p; const float* b1;
  const _Float16* W2p; const float* b2;   // W2p: KT=2 pack, cols 0..2
  float* sh_sum;
};

__global__ __launch_bounds__(256) void shinit_kernel(ShInitArgs A) {
  __shared__ _Float16 Xs[32 * 160];
  __shared__ _Float16 Hs[32 * 64];
  __shared__ int rows[32], cols[32];
  __shared__ float radS[32][16], sheS[32][9], g0S[32][3];
  const int tid = threadIdx.x, lane = tid & 31, wave = tid >> 5;
  const int eb = blockIdx.x * 32;
  const int rowBase = (wave >> 2) * 16, colBase = (wave & 3) * 16;

  if (tid < 32) {
    int e = eb + tid;
    int r = A.eidx[e], c = A.eidx[kE + e];
    rows[tid] = r; cols[tid] = c;
    float rx = A.relv[e * 3], ry = A.relv[e * 3 + 1], rz = A.relv[e * 3 + 2];
    float rr = sqrtf(rx * rx + ry * ry + rz * rz);
    radial_basis(rr, radS[tid]);
    float inv = 1.0f / (rr + kEPS);
    sh_basis(rx * inv, ry * inv, rz * inv, sheS[tid]);
  }
  __syncthreads();
  {  // h gathers: 16-byte chunks, f16 end-to-end
    int e = tid >> 3, q = tid & 7;
    const v8h* hr = (const v8h*)(A.h16 + (size_t)rows[e] * 64);
    const v8h* hc = (const v8h*)(A.h16 + (size_t)cols[e] * 64);
    *(v8h*)&Xs[e * 160 + q * 8]      = hr[q];
    *(v8h*)&Xs[e * 160 + 64 + q * 8] = hc[q];
  }
  for (int idx = tid; idx < 32 * 32; idx += 256) {
    int e = idx >> 5, q = idx & 31;
    Xs[e * 160 + 128 + q] = (q < 16) ? (_Float16)radS[e][q] : (_Float16)0.0f;
  }
  __syncthreads();
  { v8f c = wave_gemm_f16<5>(Xs, 160, rowBase, A.W1p, colBase, lane);
    store_silu_lds(Hs, 64, rowBase, colBase, lane, c, A.b1); }
  __syncthreads();
  if ((wave & 3) == 0) {  // waves 0 & 4: 64x3 output head as WMMA
    v8f c = wave_gemm_f16<2>(Hs, 64, rowBase, A.W2p, 0, lane);
    const int n = lane & 15, mo = (lane & 16) ? 8 : 0;
    if (n < 3) {
      const float bv = A.b2[n];
#pragma unroll
      for (int v = 0; v < 8; ++v) g0S[rowBase + mo + v][n] = c[v] + bv;
    }
  }
  __syncthreads();
  for (int idx = tid; idx < 32 * 9; idx += 256) {
    int e = idx / 9, j = idx % 9;
    int g = (j == 0) ? 0 : ((j < 4) ? 1 : 2);
    atomicAdd(&A.sh_sum[(size_t)rows[e] * 9 + j], g0S[e][g] * sheS[e][j]);
  }
}

// ---------------------------------------------------------------------------
// Fused per-layer edge kernel: msg MLP (147->64->64), pos/vel/sh gate MLPs via
// a single block-diagonal 192x16 WMMA head, equivariant products, scatter-adds.
// ---------------------------------------------------------------------------
struct EdgeLayerArgs {
  const int* eidx; const _Float16* h16; const float* node_sh;
  const float* relv; const float* dvelv;
  const _Float16 *msgW1p, *msgW2p, *posW1p, *velW1p, *shW1p, *gateW2p;
  const float *msg_b1, *msg_b2;
  const float *pos_b1, *pos_b2;
  const float *vel_b1, *vel_b2;
  const float *sh_b1,  *sh_b2;
  float *msg_sum, *pos_sum, *vel_sum, *sh_sum;
};

__global__ __launch_bounds__(256) void edge_layer_kernel(EdgeLayerArgs A) {
  __shared__ _Float16 Xs[32 * 160];   // input staging; later reused as Hp/Hv
  __shared__ _Float16 Hs[32 * 64];    // msg hidden; later reused as Hsh
  __shared__ _Float16 Ms[32 * 64];    // msg output
  __shared__ int rows[32], cols[32];
  __shared__ float relS[32][3], dvS[32][3];
  __shared__ float shrS[32][9], shcS[32][9];
  __shared__ float radS[32][16], ipS[32][3];
  __shared__ float pgS[32][2], vgS[32][2], wgS[32][3];
  const int tid = threadIdx.x, lane = tid & 31, wave = tid >> 5;
  const int eb = blockIdx.x * 32;
  const int rowBase = (wave >> 2) * 16, colBase = (wave & 3) * 16;
  _Float16* Hp  = Xs;          // 32x64 (Xs is free after msg GEMM1)
  _Float16* Hv  = Xs + 2048;   // 32x64
  _Float16* Hsh = Hs;          // 32x64 (Hs is free after msg GEMM2)

  if (tid < 32) {
    int e = eb + tid;
    int r = A.eidx[e], c = A.eidx[kE + e];
    rows[tid] = r; cols[tid] = c;
    float rx = A.relv[e * 3], ry = A.relv[e * 3 + 1], rz = A.relv[e * 3 + 2];
    relS[tid][0] = rx; relS[tid][1] = ry; relS[tid][2] = rz;
    dvS[tid][0] = A.dvelv[e * 3]; dvS[tid][1] = A.dvelv[e * 3 + 1]; dvS[tid][2] = A.dvelv[e * 3 + 2];
    radial_basis(sqrtf(rx * rx + ry * ry + rz * rz), radS[tid]);
    float ip0 = 0.f, ip1 = 0.f, ip2 = 0.f;
#pragma unroll
    for (int j = 0; j < 9; ++j) {
      float a = A.node_sh[(size_t)r * 9 + j], b = A.node_sh[(size_t)c * 9 + j];
      shrS[tid][j] = a; shcS[tid][j] = b;
      float p = a * b;
      if (j == 0) ip0 += p; else if (j < 4) ip1 += p; else ip2 += p;
    }
    ipS[tid][0] = ip0; ipS[tid][1] = ip1; ipS[tid][2] = ip2;
  }
  __syncthreads();
  {  // h gathers: one 16B chunk per thread per side
    int e = tid >> 3, q = tid & 7;
    const v8h* hr = (const v8h*)(A.h16 + (size_t)rows[e] * 64);
    const v8h* hc = (const v8h*)(A.h16 + (size_t)cols[e] * 64);
    *(v8h*)&Xs[e * 160 + q * 8]      = hr[q];
    *(v8h*)&Xs[e * 160 + 64 + q * 8] = hc[q];
  }
  for (int idx = tid; idx < 32 * 32; idx += 256) {
    int e = idx >> 5, q = idx & 31;
    float v = (q < 16) ? radS[e][q] : ((q < 19) ? ipS[e][q - 16] : 0.0f);
    Xs[e * 160 + 128 + q] = (_Float16)v;
  }
  __syncthreads();
  // msg MLP
  { v8f c = wave_gemm_f16<5>(Xs, 160, rowBase, A.msgW1p, colBase, lane);
    store_silu_lds(Hs, 64, rowBase, colBase, lane, c, A.msg_b1); }
  __syncthreads();
  { v8f c = wave_gemm_f16<2>(Hs, 64, rowBase, A.msgW2p, colBase, lane);
    store_silu_lds(Ms, 64, rowBase, colBase, lane, c, A.msg_b2); }
  __syncthreads();
  // msg scatter + three gate hidden layers (disjoint buffers, no barriers between)
  for (int idx = tid; idx < 32 * 64; idx += 256) {
    int e = idx >> 6, cc = idx & 63;
    atomicAdd(&A.msg_sum[(size_t)rows[e] * 64 + cc], (float)Ms[e * 64 + cc]);
  }
  { v8f c = wave_gemm_f16<2>(Ms, 64, rowBase, A.posW1p, colBase, lane);
    store_silu_lds(Hp, 64, rowBase, colBase, lane, c, A.pos_b1); }
  { v8f c = wave_gemm_f16<2>(Ms, 64, rowBase, A.velW1p, colBase, lane);
    store_silu_lds(Hv, 64, rowBase, colBase, lane, c, A.vel_b1); }
  { v8f c = wave_gemm_f16<2>(Ms, 64, rowBase, A.shW1p, colBase, lane);
    store_silu_lds(Hsh, 64, rowBase, colBase, lane, c, A.sh_b1); }
  __syncthreads();
  // all seven gate outputs in one block-diagonal WMMA head (waves 0 & 4)
  if ((wave & 3) == 0) {
    v8f c = {};
    c = wave_gemm_f16_acc<2>(c, Hp, 64, rowBase, A.gateW2p, 0, 0, lane);
    c = wave_gemm_f16_acc<2>(c, Hv, 64, rowBase, A.gateW2p, 2, 0, lane);
    c = wave_gemm_f16_acc<2>(c, Hsh, 64, rowBase, A.gateW2p, 4, 0, lane);
    const int n = lane & 15, mo = (lane & 16) ? 8 : 0;
    if (n < 7) {
#pragma unroll
      for (int v = 0; v < 8; ++v) {
        int e = rowBase + mo + v;
        float val = c[v];
        if (n < 2)      pgS[e][n]     = val + A.pos_b2[n];
        else if (n < 4) vgS[e][n - 2] = val + A.vel_b2[n - 2];
        else            wgS[e][n - 4] = val + A.sh_b2[n - 4];
      }
    }
  }
  __syncthreads();
  // equivariant products + scatter
  for (int idx = tid; idx < 32 * 15; idx += 256) {
    int e = idx / 15, q = idx % 15;
    int r = rows[e];
    if (q < 3) {
      atomicAdd(&A.pos_sum[(size_t)r * 3 + q], pgS[e][0] * relS[e][q] + pgS[e][1] * dvS[e][q]);
    } else if (q < 6) {
      int k2 = q - 3;
      atomicAdd(&A.vel_sum[(size_t)r * 3 + k2], vgS[e][0] * dvS[e][k2] + vgS[e][1] * relS[e][k2]);
    } else {
      int j = q - 6;
      int g = (j == 0) ? 0 : ((j < 4) ? 1 : 2);
      atomicAdd(&A.sh_sum[(size_t)r * 9 + j], wgS[e][g] * (shrS[e][j] - shcS[e][j]));
    }
  }
}

// ---------------------------------------------------------------------------
// Node update: h' = MLP(128->64->64)(h, msg_mean); node_sh/delta accumulation
// ---------------------------------------------------------------------------
struct NodeUpdArgs {
  const _Float16* h16_in; _Float16* h16_out;
  const _Float16* W1p; const _Float16* W2p;
  const float* b1; const float* b2;
  const float *msg_sum, *pos_sum, *vel_sum, *sh_sum, *deg;
  float *node_sh, *dpos, *dvel;
};

__global__ __launch_bounds__(256) void node_update_kernel(NodeUpdArgs A) {
  __shared__ _Float16 Xs[32 * 128];
  __shared__ _Float16 Hs[32 * 64];
  const int tid = threadIdx.x, lane = tid & 31, wave = tid >> 5;
  const int nb = blockIdx.x * 32;
  const int rowBase = (wave >> 2) * 16, colBase = (wave & 3) * 16;

  {
    int e = tid >> 3, q = tid & 7;
    *(v8h*)&Xs[e * 128 + q * 8] = ((const v8h*)(A.h16_in + (size_t)(nb + e) * 64))[q];
  }
  for (int idx = tid; idx < 32 * 64; idx += 256) {
    int e = idx >> 6, cc = idx & 63;
    int n = nb + e;
    float dg = fmaxf(A.deg[n], 1.0f);
    Xs[e * 128 + 64 + cc] = (_Float16)(A.msg_sum[(size_t)n * 64 + cc] / dg);
  }
  __syncthreads();
  { v8f c = wave_gemm_f16<4>(Xs, 128, rowBase, A.W1p, colBase, lane);
    store_silu_lds(Hs, 64, rowBase, colBase, lane, c, A.b1); }
  __syncthreads();
  {
    v8f c = wave_gemm_f16<2>(Hs, 64, rowBase, A.W2p, colBase, lane);
    const int n = colBase + (lane & 15);
    const int mo = (lane & 16) ? 8 : 0;
    const float bv = A.b2[n];
#pragma unroll
    for (int v = 0; v < 8; ++v)
      A.h16_out[(size_t)(nb + rowBase + mo + v) * 64 + n] = (_Float16)(c[v] + bv);
  }
  for (int idx = tid; idx < 32 * 15; idx += 256) {
    int e = idx / 15, q = idx % 15;
    int n = nb + e;
    float dg = fmaxf(A.deg[n], 1.0f);
    if (q < 3)      A.dpos[n * 3 + q]          += A.pos_sum[n * 3 + q] / dg;
    else if (q < 6) A.dvel[n * 3 + (q - 3)]    += A.vel_sum[n * 3 + (q - 3)] / dg;
    else            A.node_sh[n * 9 + (q - 6)] += A.sh_sum[n * 9 + (q - 6)] / dg;
  }
}

// ---------------------------------------------------------------------------
// Output heads: pos_head (67->64->3) and vel_head (70->64->3)
// ---------------------------------------------------------------------------
struct HeadsArgs {
  const _Float16* h16; const float *dpos, *dvel, *vel;
  const _Float16* pW1p; const float* p_b1; const _Float16* pW2p; const float* p_b2;
  const _Float16* vW1p; const float* v_b1; const _Float16* vW2p; const float* v_b2;
  float* out;
};

__global__ __launch_bounds__(256) void heads_kernel(HeadsArgs A) {
  __shared__ _Float16 Xs[32 * 96];
  __shared__ _Float16 Hs[32 * 64];
  const int tid = threadIdx.x, lane = tid & 31, wave = tid >> 5;
  const int nb = blockIdx.x * 32;
  const int rowBase = (wave >> 2) * 16, colBase = (wave & 3) * 16;

  {  // h columns staged once, reused by both heads
    int e = tid >> 3, q = tid & 7;
    *(v8h*)&Xs[e * 96 + q * 8] = ((const v8h*)(A.h16 + (size_t)(nb + e) * 64))[q];
  }
  for (int idx = tid; idx < 32 * 32; idx += 256) {
    int e = idx >> 5, q = idx & 31;
    int n = nb + e;
    Xs[e * 96 + 64 + q] = (q < 3) ? (_Float16)A.dpos[n * 3 + q] : (_Float16)0.0f;
  }
  __syncthreads();
  { v8f c = wave_gemm_f16<3>(Xs, 96, rowBase, A.pW1p, colBase, lane);
    store_silu_lds(Hs, 64, rowBase, colBase, lane, c, A.p_b1); }
  __syncthreads();
  if ((wave & 3) == 0) {  // pos output head via WMMA, straight to global
    v8f c = wave_gemm_f16<2>(Hs, 64, rowBase, A.pW2p, 0, lane);
    const int n = lane & 15, mo = (lane & 16) ? 8 : 0;
    if (n < 3) {
      const float bv = A.p_b2[n];
#pragma unroll
      for (int v = 0; v < 8; ++v)
        A.out[(size_t)(nb + rowBase + mo + v) * 6 + n] = c[v] + bv;
    }
  }
  for (int idx = tid; idx < 32 * 32; idx += 256) {  // only tail columns change
    int e = idx >> 5, q = idx & 31;
    int n = nb + e;
    float v = 0.0f;
    if (q < 3) v = A.dvel[n * 3 + q];
    else if (q < 6) v = A.vel[n * 3 + (q - 3)];
    Xs[e * 96 + 64 + q] = (_Float16)v;
  }
  __syncthreads();
  { v8f c = wave_gemm_f16<3>(Xs, 96, rowBase, A.vW1p, colBase, lane);
    store_silu_lds(Hs, 64, rowBase, colBase, lane, c, A.v_b1); }
  __syncthreads();
  if ((wave & 3) == 0) {  // vel output head
    v8f c = wave_gemm_f16<2>(Hs, 64, rowBase, A.vW2p, 0, lane);
    const int n = lane & 15, mo = (lane & 16) ? 8 : 0;
    if (n < 3) {
      const float bv = A.v_b2[n];
#pragma unroll
      for (int v = 0; v < 8; ++v)
        A.out[(size_t)(nb + rowBase + mo + v) * 6 + 3 + n] = c[v] + bv;
    }
  }
}

// ---------------------------------------------------------------------------
// Host driver
// ---------------------------------------------------------------------------
struct MLPIdx { int W1, b1, W2, b2; };

extern "C" void kernel_launch(void* const* d_in, const int* in_sizes, int n_in,
                              void* d_out, int out_size, void* d_ws, size_t ws_size,
                              hipStream_t stream) {
  (void)out_size; (void)ws_size;

  // ---- param leaf index map (detect flatten ordering) ----
  int embW = 4, embb = 5;
  MLPIdx shi, ph, vh, msg[4], posm[4], velm[4], nodem[4], shm[4];
  const bool sortedOrder = (n_in > 6 && in_sizes[6] == 147 * 64);
  if (!sortedOrder) {  // recursive dict-insertion order
    shi = {6, 7, 8, 9}; ph = {10, 11, 12, 13}; vh = {14, 15, 16, 17};
    for (int l = 0; l < 4; ++l) {
      int b = 18 + 20 * l;
      msg[l]   = {b + 0,  b + 1,  b + 2,  b + 3};
      posm[l]  = {b + 4,  b + 5,  b + 6,  b + 7};
      velm[l]  = {b + 8,  b + 9,  b + 10, b + 11};
      nodem[l] = {b + 12, b + 13, b + 14, b + 15};
      shm[l]   = {b + 16, b + 17, b + 18, b + 19};
    }
  } else {  // jax pytree sorted-key order
    for (int l = 0; l < 4; ++l) {
      int b = 6 + 20 * l;
      msg[l]   = {b + 0,  b + 2,  b + 1,  b + 3};
      nodem[l] = {b + 4,  b + 6,  b + 5,  b + 7};
      posm[l]  = {b + 8,  b + 10, b + 9,  b + 11};
      shm[l]   = {b + 12, b + 14, b + 13, b + 15};
      velm[l]  = {b + 16, b + 18, b + 17, b + 19};
    }
    ph = {86, 88, 87, 89}; shi = {90, 92, 91, 93}; vh = {94, 96, 95, 97};
  }
  auto F = [&](int i) { return (const float*)d_in[i]; };

  // ---- workspace carve ----
  char* base = (char*)d_ws;
  size_t off = 0;
  auto alloc = [&](size_t bytes) -> char* {
    char* p = base + off;
    off = (off + bytes + 255) & ~(size_t)255;
    return p;
  };
  _Float16* h16a = (_Float16*)alloc((size_t)kN * 64 * 2);
  _Float16* h16b = (_Float16*)alloc((size_t)kN * 64 * 2);
  float* node_sh = (float*)alloc((size_t)kN * 9 * 4);
  float* deg     = (float*)alloc((size_t)kN * 4);
  float* dpos    = (float*)alloc((size_t)kN * 6 * 4);
  float* dvel    = dpos + (size_t)kN * 3;
  float* acc     = (float*)alloc((size_t)kN * 79 * 4);
  float* msg_sum = acc;
  float* pos_sum = acc + (size_t)kN * 64;
  float* vel_sum = pos_sum + (size_t)kN * 3;
  float* sh_sum  = vel_sum + (size_t)kN * 3;
  float* relv    = (float*)alloc((size_t)kE * 3 * 4);
  float* dvelv   = (float*)alloc((size_t)kE * 3 * 4);
  _Float16* shiW1p = (_Float16*)alloc(5 * 2048 * 2);
  _Float16* shiW2p = (_Float16*)alloc(2 * 2048 * 2);
  _Float16 *msgW1p[4], *msgW2p[4], *nodeW1p[4], *nodeW2p[4], *posW1p[4], *velW1p[4], *shW1p[4],
           *gateW2p[4];
  for (int l = 0; l < 4; ++l) {
    msgW1p[l]  = (_Float16*)alloc(5 * 2048 * 2);
    msgW2p[l]  = (_Float16*)alloc(2 * 2048 * 2);
    nodeW1p[l] = (_Float16*)alloc(4 * 2048 * 2);
    nodeW2p[l] = (_Float16*)alloc(2 * 2048 * 2);
    posW1p[l]  = (_Float16*)alloc(2 * 2048 * 2);
    velW1p[l]  = (_Float16*)alloc(2 * 2048 * 2);
    shW1p[l]   = (_Float16*)alloc(2 * 2048 * 2);
    gateW2p[l] = (_Float16*)alloc(6 * 2048 * 2);
  }
  _Float16* phW1p = (_Float16*)alloc(3 * 2048 * 2);
  _Float16* phW2p = (_Float16*)alloc(2 * 2048 * 2);
  _Float16* vhW1p = (_Float16*)alloc(3 * 2048 * 2);
  _Float16* vhW2p = (_Float16*)alloc(2 * 2048 * 2);

  // ---- pack weights to fragment-swizzled f16 ----
  auto convW = [&](int leaf, int K, int KT, _Float16* dst) {
    int tot = KT * 2048;
    convert_w_kernel<<<(tot + 255) / 256, 256, 0, stream>>>(F(leaf), dst, K, KT);
  };
  auto convS = [&](int leaf, int ncols, _Float16* dst) {
    convert_small_w_kernel<<<(2 * 2048 + 255) / 256, 256, 0, stream>>>(F(leaf), dst, ncols);
  };
  convW(shi.W1, 144, 5, shiW1p);
  convS(shi.W2, 3, shiW2p);
  for (int l = 0; l < 4; ++l) {
    convW(msg[l].W1, 147, 5, msgW1p[l]);
    convW(msg[l].W2, 64, 2, msgW2p[l]);
    convW(nodem[l].W1, 128, 4, nodeW1p[l]);
    convW(nodem[l].W2, 64, 2, nodeW2p[l]);
    convW(posm[l].W1, 64, 2, posW1p[l]);
    convW(velm[l].W1, 64, 2, velW1p[l]);
    convW(shm[l].W1, 64, 2, shW1p[l]);
    convert_gate_w_kernel<<<(6 * 2048 + 255) / 256, 256, 0, stream>>>(
        F(posm[l].W2), F(velm[l].W2), F(shm[l].W2), gateW2p[l]);
  }
  convW(ph.W1, 67, 3, phW1p);
  convS(ph.W2, 3, phW2p);
  convW(vh.W1, 70, 3, vhW1p);
  convS(vh.W2, 3, vhW2p);

  // ---- init ----
  zero_kernel<<<(kN + 255) / 256, 256, 0, stream>>>(deg, kN);
  zero_kernel<<<(kN * 79 + 255) / 256, 256, 0, stream>>>(acc, kN * 79);
  zero_kernel<<<(kN * 6 + 255) / 256, 256, 0, stream>>>(dpos, kN * 6);

  const int* eidx = (const int*)d_in[3];
  embed_kernel<<<(kN * 64 + 255) / 256, 256, 0, stream>>>(F(0), F(embW), F(embb), h16a);
  edge_geom_kernel<<<(kE + 255) / 256, 256, 0, stream>>>(F(1), F(2), eidx, relv, dvelv, deg);

  // ---- sh_init ----
  {
    ShInitArgs a;
    a.eidx = eidx; a.h16 = h16a; a.relv = relv;
    a.W1p = shiW1p; a.b1 = F(shi.b1); a.W2p = shiW2p; a.b2 = F(shi.b2);
    a.sh_sum = sh_sum;
    shinit_kernel<<<kE / 32, 256, 0, stream>>>(a);
  }
  finalize_sh_kernel<<<(kN * 9 + 255) / 256, 256, 0, stream>>>(sh_sum, deg, node_sh);

  // ---- message passing layers ----
  _Float16* hcur = h16a;
  _Float16* hnxt = h16b;
  for (int l = 0; l < 4; ++l) {
    zero_kernel<<<(kN * 79 + 255) / 256, 256, 0, stream>>>(acc, kN * 79);
    {
      EdgeLayerArgs a;
      a.eidx = eidx; a.h16 = hcur; a.node_sh = node_sh; a.relv = relv; a.dvelv = dvelv;
      a.msgW1p = msgW1p[l]; a.msgW2p = msgW2p[l];
      a.posW1p = posW1p[l]; a.velW1p = velW1p[l]; a.shW1p = shW1p[l]; a.gateW2p = gateW2p[l];
      a.msg_b1 = F(msg[l].b1); a.msg_b2 = F(msg[l].b2);
      a.pos_b1 = F(posm[l].b1); a.pos_b2 = F(posm[l].b2);
      a.vel_b1 = F(velm[l].b1); a.vel_b2 = F(velm[l].b2);
      a.sh_b1  = F(shm[l].b1);  a.sh_b2  = F(shm[l].b2);
      a.msg_sum = msg_sum; a.pos_sum = pos_sum; a.vel_sum = vel_sum; a.sh_sum = sh_sum;
      edge_layer_kernel<<<kE / 32, 256, 0, stream>>>(a);
    }
    {
      NodeUpdArgs a;
      a.h16_in = hcur; a.h16_out = hnxt;
      a.W1p = nodeW1p[l]; a.W2p = nodeW2p[l];
      a.b1 = F(nodem[l].b1); a.b2 = F(nodem[l].b2);
      a.msg_sum = msg_sum; a.pos_sum = pos_sum; a.vel_sum = vel_sum; a.sh_sum = sh_sum;
      a.deg = deg; a.node_sh = node_sh; a.dpos = dpos; a.dvel = dvel;
      node_update_kernel<<<kN / 32, 256, 0, stream>>>(a);
    }
    _Float16* t = hcur; hcur = hnxt; hnxt = t;
  }

  // ---- output heads ----
  {
    HeadsArgs a;
    a.h16 = hcur; a.dpos = dpos; a.dvel = dvel; a.vel = F(2);
    a.pW1p = phW1p; a.p_b1 = F(ph.b1); a.pW2p = phW2p; a.p_b2 = F(ph.b2);
    a.vW1p = vhW1p; a.v_b1 = F(vh.b1); a.vW2p = vhW2p; a.v_b2 = F(vh.b2);
    a.out = (float*)d_out;
    heads_kernel<<<kN / 32, 256, 0, stream>>>(a);
  }
}
